// SimpleGAT_58445914964186
// MI455X (gfx1250) — matrix operand order repaired and checked by reading, not compiled, not run on hardware
//
#include <hip/hip_runtime.h>

typedef __attribute__((ext_vector_type(16))) _Float16 v16h;
typedef __attribute__((ext_vector_type(8)))  _Float16 v8h;
typedef __attribute__((ext_vector_type(8)))  float    v8f;

#define NEG_SLOPE 0.2f

// ---------------- utility kernels ----------------

__global__ void k_f32_to_f16(const float* __restrict__ in, _Float16* __restrict__ out, int n) {
    int i = blockIdx.x * blockDim.x + threadIdx.x;
    if (i < n) out[i] = (_Float16)in[i];
}

// Pack W [K, N] (f32, row-major) into per-lane WMMA B layout (f16):
// Bp[((nt*ksteps + kb)*32 + lane)*16 + e] = W[kb*32 + (e<8 ? (lane>>4)*8 + e : 16 + (lane>>4)*8 + e-8), nt*16 + (lane&15)]
// (16-bit B 32x16: half-wave h covers K 8h..8h+7 and 16+8h..16+8h+7; N = lane&15)
__global__ void k_pack_b(const float* __restrict__ W, _Float16* __restrict__ Bp, int K, int N) {
    int idx = blockIdx.x * blockDim.x + threadIdx.x;
    int total = (N >> 4) * (K >> 5) * 512;
    if (idx >= total) return;
    int e    = idx & 15;
    int lane = (idx >> 4) & 31;
    int rest = idx >> 9;               // nt*ksteps + kb
    int ksteps = K >> 5;
    int kb = rest % ksteps;
    int nt = rest / ksteps;
    int k = kb * 32 + (e < 8 ? (lane >> 4) * 8 + e : 16 + (lane >> 4) * 8 + (e - 8));
    int n = nt * 16 + (lane & 15);
    Bp[idx] = (_Float16)W[(size_t)k * N + n];
}

// -------- WMMA GEMM: C[M,N] = A[M,K](f16) * Bp(packed f16), f32 accum --------
// Block = 8 waves; each wave owns a 32x64 output (2 M-tiles x 4 N-tiles) so every
// B fragment fetched from LDS feeds TWO v_wmma ops (8 WMMA per K-step per wave).
// The block's B panel is contiguous in the packed layout (4 tiles * ksteps * 1KB)
// and staged once into LDS (<=64KB, well under the 320KB/WGP budget): B reads are
// ds_load_b128 hits, B global traffic is 1 load per block instead of per wave.
__global__ __launch_bounds__(256)
void k_gemm_wmma(const _Float16* __restrict__ A, const _Float16* __restrict__ Bp,
                 float* __restrict__ C, int M, int K, int N) {
    extern __shared__ v16h ldsB[];          // [4 * ksteps * 32] v16h (32B aligned)
    const int lane   = threadIdx.x & 31;
    const int wave   = threadIdx.x >> 5;
    const int ksteps = K >> 5;
    const int nt0    = blockIdx.x * 4;

    // cooperative stage of the contiguous B panel into LDS
    const int panel = 4 * ksteps * 32;      // in v16h units
    const v16h* bsrc = (const v16h*)Bp + (size_t)nt0 * ksteps * 32;
    for (int i = threadIdx.x; i < panel; i += 256) ldsB[i] = bsrc[i];
    __syncthreads();

    const int mt = blockIdx.y * 8 + wave;   // 32-row tile index
    if (mt * 32 >= M) return;               // after the barrier: no divergence hazard
    const int m0   = mt * 32 + (lane & 15); // A row, M-tile 0
    const int koff = (lane >> 4) * 8;       // A 16-bit layout: half-wave K split
    const _Float16* arow0 = A + (size_t)m0 * K;
    const _Float16* arow1 = arow0 + (size_t)16 * K;   // M-tile 1 (rows +16)

    v8f acc00 = {}, acc01 = {}, acc02 = {}, acc03 = {};   // M-tile 0, N-tiles 0..3
    v8f acc10 = {}, acc11 = {}, acc12 = {}, acc13 = {};   // M-tile 1, N-tiles 0..3
    for (int kb = 0; kb < ksteps; ++kb) {
        const int ko = kb * 32;
        if (kb + 1 < ksteps) {               // WGP-scope prefetch of next A slices
            __builtin_prefetch(arow0 + ko + 32, 0, 3);
            __builtin_prefetch(arow1 + ko + 32, 0, 3);
        }
        v16h a0, a1;
        {
            v8h lo = *(const v8h*)(arow0 + ko + koff);       // K = koff..koff+7
            v8h hi = *(const v8h*)(arow0 + ko + 16 + koff);  // K = 16+koff..
#pragma unroll
            for (int i = 0; i < 8; ++i) { a0[i] = lo[i]; a0[8 + i] = hi[i]; }
        }
        {
            v8h lo = *(const v8h*)(arow1 + ko + koff);
            v8h hi = *(const v8h*)(arow1 + ko + 16 + koff);
#pragma unroll
            for (int i = 0; i < 8; ++i) { a1[i] = lo[i]; a1[8 + i] = hi[i]; }
        }
        v16h b;
        b = ldsB[(0 * ksteps + kb) * 32 + lane];
        acc00 = __builtin_amdgcn_wmma_f32_16x16x32_f16(false, a0, false, b, (short)0, acc00, false, false);
        acc10 = __builtin_amdgcn_wmma_f32_16x16x32_f16(false, a1, false, b, (short)0, acc10, false, false);
        b = ldsB[(1 * ksteps + kb) * 32 + lane];
        acc01 = __builtin_amdgcn_wmma_f32_16x16x32_f16(false, a0, false, b, (short)0, acc01, false, false);
        acc11 = __builtin_amdgcn_wmma_f32_16x16x32_f16(false, a1, false, b, (short)0, acc11, false, false);
        b = ldsB[(2 * ksteps + kb) * 32 + lane];
        acc02 = __builtin_amdgcn_wmma_f32_16x16x32_f16(false, a0, false, b, (short)0, acc02, false, false);
        acc12 = __builtin_amdgcn_wmma_f32_16x16x32_f16(false, a1, false, b, (short)0, acc12, false, false);
        b = ldsB[(3 * ksteps + kb) * 32 + lane];
        acc03 = __builtin_amdgcn_wmma_f32_16x16x32_f16(false, a0, false, b, (short)0, acc03, false, false);
        acc13 = __builtin_amdgcn_wmma_f32_16x16x32_f16(false, a1, false, b, (short)0, acc13, false, false);
    }
    // C layout: VGPR r -> row (lane<16 ? r : 8+r), col = lane&15
    const int mrow = mt * 32 + (lane >> 4) * 8;
    const int nc   = nt0 * 16 + (lane & 15);
#pragma unroll
    for (int r = 0; r < 8; ++r) {
        float* crow = C + (size_t)(mrow + r) * N + nc;
        crow[0]  = acc00[r];
        crow[16] = acc01[r];
        crow[32] = acc02[r];
        crow[48] = acc03[r];
        float* crow1 = crow + (size_t)16 * N;
        crow1[0]  = acc10[r];
        crow1[16] = acc11[r];
        crow1[32] = acc12[r];
        crow1[48] = acc13[r];
    }
}

// ---------------- per-node attention logits: warp per (node, head) ----------------
__global__ void k_alpha(const float* __restrict__ xw, const float* __restrict__ a_src,
                        const float* __restrict__ a_dst, float* __restrict__ as,
                        float* __restrict__ ad, int Nt, int H, int C) {
    int w    = (blockIdx.x * blockDim.x + threadIdx.x) >> 5;
    int lane = threadIdx.x & 31;
    if (w >= Nt * H) return;
    int n = w / H, h = w % H;
    const float* row = xw + (size_t)n * H * C + h * C;
    const float* s = a_src + h * C;
    const float* d = a_dst + h * C;
    float ss = 0.f, dd = 0.f;
    for (int c = lane; c < C; c += 32) { float v = row[c]; ss += v * s[c]; dd += v * d[c]; }
#pragma unroll
    for (int off = 16; off; off >>= 1) {
        ss += __shfl_down(ss, off, 32);
        dd += __shfl_down(dd, off, 32);
    }
    if (lane == 0) { as[w] = ss; ad[w] = dd; }
}

// ---------------- edge helpers ----------------
__device__ __forceinline__ void edge_sd(const int* __restrict__ ei, int E, int N,
                                        int t, int& s, int& d) {
    if (t < 2 * E) {
        int b = t / E;
        int j = t - b * E;
        s = ei[j]     + b * N;
        d = ei[E + j] + b * N;
    } else {
        s = d = t - 2 * E;   // self loop
    }
}

__device__ __forceinline__ unsigned enc_f(float f) {
    unsigned u = __float_as_uint(f);
    return (u & 0x80000000u) ? ~u : (u | 0x80000000u);
}
__device__ __forceinline__ float dec_f(unsigned u) {
    return (u & 0x80000000u) ? __uint_as_float(u ^ 0x80000000u) : __uint_as_float(~u);
}

// pass 1: e = leaky_relu(as[src]+ad[dst]); atomic segment max over dst
__global__ void k_edge_max(const int* __restrict__ ei, int E, int N, int H,
                           const float* __restrict__ as, const float* __restrict__ ad,
                           float* __restrict__ ebuf, unsigned* __restrict__ menc, int Etot) {
    int idx = blockIdx.x * blockDim.x + threadIdx.x;
    if (idx >= Etot * H) return;
    int t = idx / H, h = idx - t * H;
    int s, d; edge_sd(ei, E, N, t, s, d);
    float e = as[s * H + h] + ad[d * H + h];
    e = (e > 0.f) ? e : NEG_SLOPE * e;
    ebuf[idx] = e;
    atomicMax(&menc[d * H + h], enc_f(e));
}

// pass 2: ex = exp(e - m[dst]); atomic segment sum of denom; ebuf <- ex
__global__ void k_edge_exp(const int* __restrict__ ei, int E, int N, int H,
                           float* __restrict__ ebuf, const unsigned* __restrict__ menc,
                           float* __restrict__ denom, int Etot) {
    int idx = blockIdx.x * blockDim.x + threadIdx.x;
    if (idx >= Etot * H) return;
    int t = idx / H, h = idx - t * H;
    int s, d; edge_sd(ei, E, N, t, s, d);
    float mx = dec_f(menc[d * H + h]);
    float ex = expf(ebuf[idx] - mx);
    ebuf[idx] = ex;
    atomicAdd(&denom[d * H + h], ex);
}

// pass 3: out[dst] += (ex/denom[dst]) * xw[src]   (thread = (edge, head, 4-channel group))
__global__ void k_edge_agg(const int* __restrict__ ei, int E, int N, int H, int C,
                           const float* __restrict__ ebuf, const float* __restrict__ denom,
                           const float* __restrict__ xw, float* __restrict__ out, int Etot) {
    int cg  = C >> 2;
    int idx = blockIdx.x * blockDim.x + threadIdx.x;
    if (idx >= Etot * H * cg) return;
    int c4 = idx % cg;
    int th = idx / cg;
    int h  = th % H;
    int t  = th / H;
    int s, d; edge_sd(ei, E, N, t, s, d);
    float w = ebuf[t * H + h] / (denom[d * H + h] + 1e-16f);
    const float4 v = *(const float4*)(xw + (size_t)s * H * C + h * C + c4 * 4);
    float* o = out + (size_t)d * H * C + h * C + c4 * 4;
    atomicAdd(o + 0, w * v.x);
    atomicAdd(o + 1, w * v.y);
    atomicAdd(o + 2, w * v.z);
    atomicAdd(o + 3, w * v.w);
}

// bias + exact-erf GELU, emit f16 (input to layer-2 WMMA GEMM)
__global__ void k_bias_gelu_h(const float* __restrict__ in, const float* __restrict__ b,
                              _Float16* __restrict__ out, int n, int F) {
    int i = blockIdx.x * blockDim.x + threadIdx.x;
    if (i >= n) return;
    float x = in[i] + b[i % F];
    float g = 0.5f * x * (1.0f + erff(x * 0.70710678118654752440f));
    out[i] = (_Float16)g;
}

__global__ void k_bias_add(float* __restrict__ out, const float* __restrict__ b, int n, int F) {
    int i = blockIdx.x * blockDim.x + threadIdx.x;
    if (i >= n) return;
    out[i] += b[i % F];
}

// ---------------- launch ----------------
extern "C" void kernel_launch(void* const* d_in, const int* in_sizes, int n_in,
                              void* d_out, int out_size, void* d_ws, size_t ws_size,
                              hipStream_t stream) {
    const float* x    = (const float*)d_in[0];
    const int*   ei   = (const int*)  d_in[1];
    const float* W1   = (const float*)d_in[2];
    const float* asr1 = (const float*)d_in[3];
    const float* adr1 = (const float*)d_in[4];
    const float* b1   = (const float*)d_in[5];
    const float* W2   = (const float*)d_in[6];
    const float* asr2 = (const float*)d_in[7];
    const float* adr2 = (const float*)d_in[8];
    const float* b2   = (const float*)d_in[9];
    float* out = (float*)d_out;

    constexpr int Bb = 2, Nn = 10000, IN = 256, HID = 256, H = 2, OUT = 256, E = 160000;
    constexpr int Nt   = Bb * Nn;          // 20000
    constexpr int Etot = 2 * E + Nt;       // 340000 (batched edges + self loops)
    constexpr int F1   = H * HID;          // 512

    // ---- workspace carve (~117 MB total) ----
    char* base = (char*)d_ws; size_t off = 0;
    auto alloc = [&](size_t bytes) -> char* {
        char* r = base + off; off = (off + bytes + 255) & ~(size_t)255; return r;
    };
    _Float16* Xh    = (_Float16*)alloc((size_t)Nt * IN * 2);     // x in f16
    _Float16* W1p   = (_Float16*)alloc((size_t)IN * F1 * 2);     // packed B layer1
    _Float16* W2p   = (_Float16*)alloc((size_t)F1 * OUT * 2);    // packed B layer2
    float*    xw    = (float*)   alloc((size_t)Nt * F1 * 4);     // xw1, reused as xw2
    float*    agg1  = (float*)   alloc((size_t)Nt * F1 * 4);     // layer1 aggregation
    _Float16* hh    = (_Float16*)alloc((size_t)Nt * F1 * 2);     // gelu(h) in f16
    float*    as1   = (float*)   alloc((size_t)Nt * H * 4);
    float*    ad1   = (float*)   alloc((size_t)Nt * H * 4);
    float*    as2   = (float*)   alloc((size_t)Nt * 4);
    float*    ad2   = (float*)   alloc((size_t)Nt * 4);
    unsigned* menc  = (unsigned*)alloc((size_t)Nt * H * 4);      // reused layer2 (Nt)
    float*    denom = (float*)   alloc((size_t)Nt * H * 4);      // reused layer2 (Nt)
    float*    ebuf  = (float*)   alloc((size_t)Etot * H * 4);    // reused layer2 (Etot)
    (void)ws_size; (void)n_in; (void)in_sizes; (void)out_size;

    const int TB = 256;
    auto nb = [](long long n, int tb) { return (unsigned)((n + tb - 1) / tb); };

    // ---- precision prep ----
    k_f32_to_f16<<<nb((long long)Nt * IN, TB), TB, 0, stream>>>(x, Xh, Nt * IN);
    k_pack_b<<<nb((F1 / 16) * (IN / 32) * 512, TB), TB, 0, stream>>>(W1, W1p, IN, F1);
    k_pack_b<<<nb((OUT / 16) * (F1 / 32) * 512, TB), TB, 0, stream>>>(W2, W2p, F1, OUT);

    // ================= layer 1 =================
    {
        dim3 g(F1 / 64, (Nt / 32 + 7) / 8);                         // 32 rows per wave
        size_t shmem = (size_t)4 * (IN / 32) * 32 * sizeof(v16h);   // 32 KB
        k_gemm_wmma<<<g, TB, shmem, stream>>>(Xh, W1p, xw, Nt, IN, F1);
    }
    k_alpha<<<nb((long long)Nt * H * 32, TB), TB, 0, stream>>>(xw, asr1, adr1, as1, ad1, Nt, H, HID);

    hipMemsetAsync(menc,  0, (size_t)Nt * H * 4, stream);
    hipMemsetAsync(denom, 0, (size_t)Nt * H * 4, stream);
    hipMemsetAsync(agg1,  0, (size_t)Nt * F1 * 4, stream);
    k_edge_max<<<nb((long long)Etot * H, TB), TB, 0, stream>>>(ei, E, Nn, H, as1, ad1, ebuf, menc, Etot);
    k_edge_exp<<<nb((long long)Etot * H, TB), TB, 0, stream>>>(ei, E, Nn, H, ebuf, menc, denom, Etot);
    k_edge_agg<<<nb((long long)Etot * H * (HID / 4), TB), TB, 0, stream>>>(
        ei, E, Nn, H, HID, ebuf, denom, xw, agg1, Etot);

    k_bias_gelu_h<<<nb((long long)Nt * F1, TB), TB, 0, stream>>>(agg1, b1, hh, Nt * F1, F1);

    // ================= layer 2 =================
    {
        dim3 g(OUT / 64, (Nt / 32 + 7) / 8);
        size_t shmem = (size_t)4 * (F1 / 32) * 32 * sizeof(v16h);   // 64 KB
        k_gemm_wmma<<<g, TB, shmem, stream>>>(hh, W2p, xw /* now xw2 [Nt,OUT] */, Nt, F1, OUT);
    }
    k_alpha<<<nb((long long)Nt * 32, TB), TB, 0, stream>>>(xw, asr2, adr2, as2, ad2, Nt, 1, OUT);

    hipMemsetAsync(menc,  0, (size_t)Nt * 4, stream);
    hipMemsetAsync(denom, 0, (size_t)Nt * 4, stream);
    hipMemsetAsync(out,   0, (size_t)Nt * OUT * 4, stream);
    k_edge_max<<<nb((long long)Etot, TB), TB, 0, stream>>>(ei, E, Nn, 1, as2, ad2, ebuf, menc, Etot);
    k_edge_exp<<<nb((long long)Etot, TB), TB, 0, stream>>>(ei, E, Nn, 1, ebuf, menc, denom, Etot);
    k_edge_agg<<<nb((long long)Etot * (OUT / 4), TB), TB, 0, stream>>>(
        ei, E, Nn, 1, OUT, ebuf, denom, xw, out, Etot);

    k_bias_add<<<nb((long long)Nt * OUT, TB), TB, 0, stream>>>(out, b2, Nt * OUT, OUT);
}
